// BLSTM_32160715112781
// MI455X (gfx1250) — compile-verified, matrix operand order checked by reading, not run on hardware
//
#include <hip/hip_runtime.h>
#include <hip/hip_bf16.h>

// ---------------------------------------------------------------------------
// BLSTM language model head for MI455X (gfx1250, wave32, WMMA).
// V=32000, E=64, H=256, B=8, T=512.
// Heavy GEMMs: v_wmma_f32_16x16x32_bf16 (f32 accumulate).
// Output GEMM stages its B panel in LDS via global_load_async_to_lds_b128
// (double-buffered, one k-step ahead), waited with s_wait_asynccnt.
// ---------------------------------------------------------------------------

typedef __attribute__((ext_vector_type(16))) __bf16 v16bf;
typedef __attribute__((ext_vector_type(8)))  float  v8f;

#define WMMA_BF16(a, b, c) \
  __builtin_amdgcn_wmma_f32_16x16x32_bf16(false, (a), false, (b), (short)0, (c), false, false)

static constexpr int Vv = 32000;
static constexpr int Ee = 64;
static constexpr int Hh = 256;
static constexpr int Bb = 8;
static constexpr int Tt = 512;
static constexpr int G4H = 4 * Hh;      // 1024
static constexpr int MROWS = Tt * Bb;   // 4096
static constexpr int D2H = 2 * Hh;      // 512

// --- fragment helpers -------------------------------------------------------
// 16-bit A-matrix 16x32 layout (ISA 7.12.2): lane<16 -> row=lane, K chunks
// {half*8+0..7, half*8+16..23} where half = lane>>4.
static __device__ __forceinline__ int frag_k(int half, int e) {
  return half * 8 + (e & 7) + ((e >> 3) << 4);
}

// Load A fragment (16x32 bf16) from a row-major f32 matrix, converting.
static __device__ __forceinline__ v16bf a_frag_from_f32(const float* __restrict__ base, int ld) {
  int lane = threadIdx.x & 31;
  int half = lane >> 4, row = lane & 15;
  const float* p0 = base + (size_t)row * ld + half * 8;  // K = half*8 .. +7
  const float* p1 = p0 + 16;                             // K = half*8+16 .. +23
  v16bf a;
#pragma unroll
  for (int e = 0; e < 8; ++e) a[e] = (__bf16)p0[e];
#pragma unroll
  for (int e = 0; e < 8; ++e) a[8 + e] = (__bf16)p1[e];
  return a;
}

// Same but from a bf16 row-major matrix (global h2 or LDS h state).
static __device__ __forceinline__ v16bf a_frag_from_bf16(const __bf16* __restrict__ base, int ld) {
  int lane = threadIdx.x & 31;
  int half = lane >> 4, row = lane & 15;
  const __bf16* p0 = base + (size_t)row * ld + half * 8;
  const __bf16* p1 = p0 + 16;
  v16bf a;
#pragma unroll
  for (int e = 0; e < 8; ++e) a[e] = p0[e];
#pragma unroll
  for (int e = 0; e < 8; ++e) a[8 + e] = p1[e];
  return a;
}

// Load a pre-packed B fragment: one contiguous 32B chunk per lane.
static __device__ __forceinline__ v16bf b_frag_packed(const __bf16* __restrict__ pk,
                                                      int kt, int nt, int ntiles) {
  int lane = threadIdx.x & 31;
  return *(const v16bf*)(pk + ((size_t)(kt * ntiles + nt) * 32 + lane) * 16);
}

// Store D fragment (16x16 f32) to row-major f32 (out -> tile origin).
static __device__ __forceinline__ void store_d(float* __restrict__ out, int ld, v8f d) {
  int lane = threadIdx.x & 31;
  int half = lane >> 4, col = lane & 15;
#pragma unroll
  for (int r = 0; r < 8; ++r) out[(size_t)(r + half * 8) * ld + col] = d[r];
}

static __device__ __forceinline__ float sigm(float x) {
  return 1.0f / (1.0f + __expf(-x));
}

// --- kernel 0: pack a row-major f32 [K,N] matrix into bf16 B-fragments ------
__global__ void pack_b_kernel(const float* __restrict__ src, __bf16* __restrict__ dst,
                              int K, int N) {
  int tid = blockIdx.x * blockDim.x + threadIdx.x;
  int lane = tid & 31;
  int frag = tid >> 5;
  int ntiles = N >> 4;
  int ktiles = K >> 5;
  int kt = frag / ntiles, nt = frag % ntiles;
  if (kt >= ktiles) return;
  int half = lane >> 4;
  int col = nt * 16 + (lane & 15);
  __bf16* d = dst + ((size_t)frag * 32 + lane) * 16;
#pragma unroll
  for (int e = 0; e < 16; ++e) {
    int k = kt * 32 + frag_k(half, e);
    d[e] = (__bf16)src[(size_t)k * N + col];
  }
}

// --- kernel 1: embedding gather -> time-major xs [T*B, E] f32 ---------------
__global__ void embed_kernel(const int* __restrict__ x, const float* __restrict__ emb,
                             float* __restrict__ xs) {
  int tid = blockIdx.x * blockDim.x + threadIdx.x;  // MROWS*E threads
  if (tid >= MROWS * Ee) return;
  int e = tid & (Ee - 1);
  int row = tid >> 6;        // row = t*B + b
  int b = row & (Bb - 1);
  int t = row >> 3;
  int tok = x[b * Tt + t];
  xs[tid] = emb[(size_t)tok * Ee + e];
}

// --- kernel 2: input projection GEMM  xp = xs @ Wi + bias -------------------
__global__ __launch_bounds__(256) void xproj_kernel(
    const float* __restrict__ xs,
    const __bf16* __restrict__ wi_pk_f, const __bf16* __restrict__ wi_pk_b,
    const float* __restrict__ bias_f, const float* __restrict__ bias_b,
    float* __restrict__ xp_f, float* __restrict__ xp_b) {
  const __bf16* wpk = blockIdx.y ? wi_pk_b : wi_pk_f;
  const float* bias = blockIdx.y ? bias_b : bias_f;
  float* xp = blockIdx.y ? xp_b : xp_f;

  int wave = threadIdx.x >> 5;
  int g = blockIdx.x * 8 + wave;           // 256 mtiles * 64 ntiles = 16384 tiles
  int mt = g >> 6, nt = g & 63;
  int lane = threadIdx.x & 31;
  float bv = bias[nt * 16 + (lane & 15)];
  v8f acc = {bv, bv, bv, bv, bv, bv, bv, bv};
#pragma unroll
  for (int kt = 0; kt < 2; ++kt) {
    v16bf a = a_frag_from_f32(xs + (size_t)mt * 16 * Ee + kt * 32, Ee);
    v16bf b = b_frag_packed(wpk, kt, nt, G4H / 16);
    acc = WMMA_BF16(a, b, acc);
  }
  store_d(xp + (size_t)mt * 16 * G4H + nt * 16, G4H, acc);
}

// --- kernel 3: bidirectional LSTM scan --------------------------------------
// grid.x = 2 (direction), 1024 threads = 32 waves. h/c state + gates in LDS.
// Gate layout along 4H: [f, i, g, o]. h2 written directly as bf16 [B*T, 2H].
__global__ __launch_bounds__(1024) void lstm_scan_kernel(
    const float* __restrict__ xp_f, const float* __restrict__ xp_b,
    const __bf16* __restrict__ wh_pk_f, const __bf16* __restrict__ wh_pk_b,
    __bf16* __restrict__ h2) {
  __shared__ __bf16 h_s[16 * Hh];      // rows 8..15 stay zero (M padding)
  __shared__ float  c_s[Bb * Hh];
  __shared__ float  g_s[16 * G4H];     // gate pre-activations, 16x1024

  int dir = blockIdx.x;
  const float* xp = dir ? xp_b : xp_f;
  const __bf16* wpk = dir ? wh_pk_b : wh_pk_f;

  int tid = threadIdx.x;
  int lane = tid & 31, wave = tid >> 5;
  int col = lane & 15, half = lane >> 4;

  for (int i = tid; i < 16 * Hh; i += 1024) h_s[i] = (__bf16)0.0f;
  for (int i = tid; i < Bb * Hh; i += 1024) c_s[i] = 0.0f;
  __syncthreads();

  for (int s = 0; s < Tt; ++s) {
    int t = dir ? (Tt - 1 - s) : s;

    // phase 1: gates[16,1024] = h @ Wh + xp[t]   (xp already holds bias)
#pragma unroll
    for (int i = 0; i < 2; ++i) {
      int nt = wave * 2 + i;
      v8f acc;
#pragma unroll
      for (int r = 0; r < 8; ++r) {
        int row = r + half * 8;
        acc[r] = (row < Bb) ? xp[((size_t)t * Bb + row) * G4H + nt * 16 + col] : 0.0f;
      }
#pragma unroll
      for (int kt = 0; kt < 8; ++kt) {
        v16bf a = a_frag_from_bf16(h_s + kt * 32, Hh);
        v16bf b = b_frag_packed(wpk, kt, nt, G4H / 16);
        acc = WMMA_BF16(a, b, acc);
      }
#pragma unroll
      for (int r = 0; r < 8; ++r)
        g_s[(size_t)(r + half * 8) * G4H + nt * 16 + col] = acc[r];
    }
    __syncthreads();

    // phase 2: elementwise cell update, 2048 (b, hc) pairs over 1024 threads
#pragma unroll
    for (int i = 0; i < 2; ++i) {
      int e = tid + i * 1024;
      int b = e >> 8, hc = e & (Hh - 1);
      float fg = sigm(g_s[(size_t)b * G4H + hc]);
      float ig = sigm(g_s[(size_t)b * G4H + Hh + hc]);
      float gg = tanhf(g_s[(size_t)b * G4H + 2 * Hh + hc]);
      float og = sigm(g_s[(size_t)b * G4H + 3 * Hh + hc]);
      float c = fg * c_s[(size_t)b * Hh + hc] + ig * gg;
      float h = og * tanhf(c);
      c_s[(size_t)b * Hh + hc] = c;
      h_s[(size_t)b * Hh + hc] = (__bf16)h;
      // h2 laid out [B*T, 2H] with row = b*T + t so GEMM rows match [B,T,V].
      h2[((size_t)b * Tt + t) * D2H + dir * Hh + hc] = (__bf16)h;
    }
    __syncthreads();
  }
}

// --- kernel 4: output GEMM  logits = h2 @ W_out + b_out ---------------------
// M=4096, N=32000, K=512. Workgroup tile 512x32 (8 waves x 64 rows), shared
// 32-column B panel double-buffered in LDS via async global->LDS copies.
__global__ __launch_bounds__(256) void out_gemm_kernel(
    const __bf16* __restrict__ h2, const __bf16* __restrict__ wout_pk,
    const float* __restrict__ b_out, float* __restrict__ out) {
  const int NT = Vv / 16;                  // 2000 ntiles
  __shared__ __bf16 bpanel[2][2 * 512];    // 2 x (2 frags x 512 bf16) = 4KB

  int tid = threadIdx.x;
  int wave = tid >> 5, lane = tid & 31;
  int g = blockIdx.x;                      // 8 msup * 1000 nsup = 8000 blocks
  int msup = g / 1000, nsup = g % 1000;
  int mb = msup * 512 + wave * 64;         // this wave's 64-row stripe
  int nb = nsup * 32;
  int nt0 = nb >> 4;
  int col = lane & 15;

  v8f acc[4][2];
  float bv0 = b_out[nb + col];
  float bv1 = b_out[nb + 16 + col];
#pragma unroll
  for (int m = 0; m < 4; ++m) {
    acc[m][0] = (v8f){bv0, bv0, bv0, bv0, bv0, bv0, bv0, bv0};
    acc[m][1] = (v8f){bv1, bv1, bv1, bv1, bv1, bv1, bv1, bv1};
  }

  // Stage the 2KB B panel for k-step kt into bpanel[kt&1] with async copies:
  // 128 threads x 16B (one global_load_async_to_lds_b128 each).
  auto stage = [&](int kt) {
    if (tid < 128) {
      const char* src =
          (const char*)(wout_pk + (size_t)(kt * NT + nt0) * 512) + tid * 16;
      unsigned lds = (unsigned)(uintptr_t)(&bpanel[kt & 1][0]) + tid * 16;
      asm volatile("global_load_async_to_lds_b128 %0, %1, off"
                   :: "v"(lds), "v"(src) : "memory");
    }
  };

  stage(0);
#pragma unroll 4
  for (int kt = 0; kt < 16; ++kt) {
    asm volatile("s_wait_asynccnt 0x0" ::: "memory");
    __syncthreads();
    if (kt + 1 < 16) stage(kt + 1);  // overlap next panel copy with compute
    if (kt + 1 < 16)                 // warm next A stripe (global_prefetch_b8)
      __builtin_prefetch(h2 + (size_t)mb * D2H + (kt + 1) * 32, 0, 3);

    v16bf a[4];
#pragma unroll
    for (int m = 0; m < 4; ++m)
      a[m] = a_frag_from_bf16(h2 + (size_t)(mb + m * 16) * D2H + kt * 32, D2H);
    v16bf bf[2];
#pragma unroll
    for (int n = 0; n < 2; ++n)
      bf[n] = *(const v16bf*)&bpanel[kt & 1][n * 512 + lane * 16];
#pragma unroll
    for (int m = 0; m < 4; ++m)
#pragma unroll
      for (int n = 0; n < 2; ++n)
        acc[m][n] = WMMA_BF16(a[m], bf[n], acc[m][n]);
  }

#pragma unroll
  for (int m = 0; m < 4; ++m)
#pragma unroll
    for (int n = 0; n < 2; ++n)
      store_d(out + (size_t)(mb + m * 16) * Vv + nb + n * 16, Vv, acc[m][n]);
}

// ---------------------------------------------------------------------------
extern "C" void kernel_launch(void* const* d_in, const int* in_sizes, int n_in,
                              void* d_out, int out_size, void* d_ws, size_t ws_size,
                              hipStream_t stream) {
  const int*   x     = (const int*)d_in[0];
  const float* emb   = (const float*)d_in[1];
  const float* Wi_f  = (const float*)d_in[2];
  const float* Wh_f  = (const float*)d_in[3];
  const float* b_f   = (const float*)d_in[4];
  const float* Wi_b  = (const float*)d_in[5];
  const float* Wh_b  = (const float*)d_in[6];
  const float* b_b   = (const float*)d_in[7];
  const float* W_out = (const float*)d_in[8];
  const float* b_out = (const float*)d_in[9];
  float* out = (float*)d_out;

  // workspace carve-up (all 256B aligned)
  char* ws = (char*)d_ws;
  size_t off = 0;
  auto alloc = [&](size_t bytes) {
    void* p = ws + off;
    off = (off + bytes + 255) & ~(size_t)255;
    return p;
  };
  float*  xs      = (float*)alloc((size_t)MROWS * Ee * 4);        // 1 MB
  float*  xp_f    = (float*)alloc((size_t)MROWS * G4H * 4);       // 16 MB
  float*  xp_b    = (float*)alloc((size_t)MROWS * G4H * 4);       // 16 MB
  __bf16* h2      = (__bf16*)alloc((size_t)MROWS * D2H * 2);      // 4 MB
  __bf16* wi_pk_f = (__bf16*)alloc((size_t)Ee * G4H * 2);         // 128 KB
  __bf16* wi_pk_b = (__bf16*)alloc((size_t)Ee * G4H * 2);
  __bf16* wh_pk_f = (__bf16*)alloc((size_t)Hh * G4H * 2);         // 512 KB
  __bf16* wh_pk_b = (__bf16*)alloc((size_t)Hh * G4H * 2);
  __bf16* wout_pk = (__bf16*)alloc((size_t)D2H * Vv * 2);         // 32 MB

  // pack weights to bf16 B-fragment layout
  pack_b_kernel<<<(Ee / 32) * (G4H / 16) * 32 / 256, 256, 0, stream>>>(Wi_f, wi_pk_f, Ee, G4H);
  pack_b_kernel<<<(Ee / 32) * (G4H / 16) * 32 / 256, 256, 0, stream>>>(Wi_b, wi_pk_b, Ee, G4H);
  pack_b_kernel<<<(Hh / 32) * (G4H / 16) * 32 / 256, 256, 0, stream>>>(Wh_f, wh_pk_f, Hh, G4H);
  pack_b_kernel<<<(Hh / 32) * (G4H / 16) * 32 / 256, 256, 0, stream>>>(Wh_b, wh_pk_b, Hh, G4H);
  pack_b_kernel<<<(D2H / 32) * (Vv / 16) * 32 / 256, 256, 0, stream>>>(W_out, wout_pk, D2H, Vv);

  // embedding gather (time-major)
  embed_kernel<<<(MROWS * Ee) / 256, 256, 0, stream>>>(x, emb, xs);

  // input projections (both directions via grid.y)
  {
    dim3 grid((256 * 64) / 8, 2, 1);
    xproj_kernel<<<grid, 256, 0, stream>>>(xs, wi_pk_f, wi_pk_b, b_f, b_b, xp_f, xp_b);
  }

  // sequential bidirectional scan (one WGP per direction)
  lstm_scan_kernel<<<2, 1024, 0, stream>>>(xp_f, xp_b, wh_pk_f, wh_pk_b, h2);

  // output GEMM -> logits [B,T,V]
  out_gemm_kernel<<<8 * 1000, 256, 0, stream>>>(h2, wout_pk, b_out, out);

  (void)in_sizes; (void)n_in; (void)out_size; (void)ws_size;
}
